// VectorQuantizedVAE_74328704025274
// MI455X (gfx1250) — matrix operand order repaired and checked by reading, not compiled, not run on hardware
//
#include <hip/hip_runtime.h>
#include <hip/hip_bf16.h>

// ---------------------------------------------------------------------------
// MI455X (gfx1250) VQ-VAE forward.
// Channel contractions (conv implicit-GEMM, VQ distances) use
// v_wmma_f32_16x16x32_f16 (f16 in, f32 accumulate). Conv K-dim is processed
// one input channel at a time; weights are pre-padded in the workspace to
// [Cout][Cin][32] so the B fragment is 4x global_load_b128 with zero index
// math, and the A side is a 16-wide gather clause with 32-bit byte offsets
// off a uniform base (saddr+voffset addressing). Elementwise chains are fused
// to respect the 23.3 TB/s HBM roofline. Wave32 throughout.
// ---------------------------------------------------------------------------

typedef __attribute__((ext_vector_type(16))) _Float16 v16h;
typedef __attribute__((ext_vector_type(8)))  float    v8f;

#define LEAK   0.01f
#define COMMIT 7.0f

__device__ __forceinline__ float leaky1(float v) { return v >= 0.f ? v : LEAK * v; }

// 16-bit A-matrix 16x32, contiguous-k order used here (not the ISA table order): lane L serves row M = L&15; half = L>>4.
// element e -> K = half*16 + e
__device__ __forceinline__ int kmapA(int e, int hl) { return hl * 16 + e; }
// 16-bit B-matrix 32x16, same contiguous order as A: lanes 0-15 hold K=0..15, lanes 16-31 K=16..31.
__device__ __forceinline__ int kmapB(int e, int hl) { return hl * 16 + e; }

// Pad conv weights [Cout][Cin][27] -> [Cout][Cin][32], taps 27..31 = 0.
__global__ void k_pad_w(const float* __restrict__ w, float* __restrict__ wp, int total)
{
    int t = blockIdx.x * blockDim.x + threadIdx.x;
    if (t >= total) return;
    int tap = t & 31, rc = t >> 5;          // rc = cout*Cin + ci
    wp[t] = (tap < 27) ? w[rc * 27 + tap] : 0.f;
}

// ---------------------------------------------------------------------------
// Implicit-GEMM 3x3x3 conv, padding 1, stride 1 or 2; scalar bias added to the
// INPUT (reference: conv(x + bias), pad-after-bias => OOB taps contribute 0).
// Each wave computes a 16(voxel) x 16(cout) tile. K chunk = one input channel
// (27 taps padded to the 32-wide WMMA K in the pre-padded weight buffer).
// ---------------------------------------------------------------------------
__global__ void k_conv3d_wmma(const float* __restrict__ x, const float* __restrict__ bias,
                              const float* __restrict__ wp,   // padded [Cout][Cin][32]
                              float* __restrict__ y,
                              int Cin, int Cout,
                              int Din, int Hin, int Win,
                              int Dout, int Hout, int Wout, int stride)
{
    const int lane = threadIdx.x & 31;
    const int wv   = threadIdx.x >> 5;              // wave within block
    const int nwv  = blockDim.x >> 5;
    const int hl   = lane >> 4;
    const int mr   = lane & 15;

    const int Nsp = Dout * Hout * Wout;             // <= 2^21, int-safe
    const int m0  = (blockIdx.x * nwv + wv) * 16;
    const int n0  = blockIdx.y * 16;
    const float b = *bias;

    const int m = m0 + mr;                          // A row (voxel) this lane feeds
    const bool mval = (m < Nsp);
    int od = 0, oh = 0, ow = 0;
    if (mval) {
        od = m / (Hout * Wout);
        int r = m - od * Hout * Wout;
        oh = r / Wout; ow = r - oh * Wout;
    }
    const int idb = od * stride - 1, ihb = oh * stride - 1, iwb = ow * stride - 1;
    const int n    = n0 + (lane & 15);              // B/D column (cout) this lane feeds
    const bool nval = (n < Cout);

    const int HW  = Hin * Win;
    const int DHW = Din * HW;

    // ---- loop-invariant A gather byte-offsets + 0/1 masks ----
    unsigned aof[16]; float amk[16];
#pragma unroll
    for (int e = 0; e < 16; ++e) {
        int t  = kmapA(e, hl);                      // 0..31; taps 27..31 are K padding
        int kd = t / 9, r9 = t - kd * 9;
        int kh = r9 / 3, kw = r9 - kh * 3;
        int id = idb + kd, ih = ihb + kh, iw = iwb + kw;
        bool v = (t < 27) && mval &&
                 (unsigned)id < (unsigned)Din && (unsigned)ih < (unsigned)Hin &&
                 (unsigned)iw < (unsigned)Win;
        aof[e] = v ? (unsigned)((id * HW + ih * Win + iw) * 4) : 0u;
        amk[e] = v ? 1.f : 0.f;
    }

    const char*  xb = (const char*)x;                        // + ci*DHW*4
    const float* wc = wp + (nval ? n : 0) * (Cin * 32) + hl * 16;  // + ci*32; 64B aligned
    __builtin_prefetch(wc, 0, 1);

    v8f acc = {};
    for (int ci = 0; ci < Cin; ++ci) {
        float af[16];
#pragma unroll
        for (int e = 0; e < 16; ++e) af[e] = *(const float*)(xb + aof[e]);  // gather clause
        const float4 q0 = *reinterpret_cast<const float4*>(wc);
        const float4 q1 = *reinterpret_cast<const float4*>(wc + 4);
        const float4 q2 = *reinterpret_cast<const float4*>(wc + 8);
        const float4 q3 = *reinterpret_cast<const float4*>(wc + 12);
        v16h a, bm;
#pragma unroll
        for (int e = 0; e < 16; ++e) a[e] = (_Float16)((af[e] + b) * amk[e]);
        bm[0]  = (_Float16)q0.x; bm[1]  = (_Float16)q0.y;
        bm[2]  = (_Float16)q0.z; bm[3]  = (_Float16)q0.w;
        bm[4]  = (_Float16)q1.x; bm[5]  = (_Float16)q1.y;
        bm[6]  = (_Float16)q1.z; bm[7]  = (_Float16)q1.w;
        bm[8]  = (_Float16)q2.x; bm[9]  = (_Float16)q2.y;
        bm[10] = (_Float16)q2.z; bm[11] = (_Float16)q2.w;
        bm[12] = (_Float16)q3.x; bm[13] = (_Float16)q3.y;
        bm[14] = (_Float16)q3.z; bm[15] = (_Float16)q3.w;
        acc = __builtin_amdgcn_wmma_f32_16x16x32_f16(false, a, false, bm,
                                                     (short)0, acc, false, false);
        xb += (size_t)DHW * 4;
        wc += 32;
    }

    // ---- store: wave-uniform fast path (2x global_store_b128 per lane) ----
    if (m0 + 16 <= Nsp && n0 + 16 <= Cout) {        // Nsp % 8 == 0 for all layers
        int base = n * Nsp + m0 + hl * 8;
        float4 s0 = make_float4(acc[0], acc[1], acc[2], acc[3]);
        float4 s1 = make_float4(acc[4], acc[5], acc[6], acc[7]);
        *reinterpret_cast<float4*>(y + base)     = s0;
        *reinterpret_cast<float4*>(y + base + 4) = s1;
    } else {
#pragma unroll
        for (int v = 0; v < 8; ++v) {               // D: row M = v + hl*8, col N = lane&15
            int mm = m0 + v + hl * 8;
            if (mm < Nsp && nval) y[n * Nsp + mm] = acc[v];
        }
    }
}

// ---------------------------------------------------------------------------
// ConvTranspose3d k=4 s=2 p=1 (gather form: out[o] <- x[(o+1-k)/2], k parity
// fixed by o).  Tiny layers -> direct VALU.
// ---------------------------------------------------------------------------
__global__ void k_convT3d(const float* __restrict__ x, const float* __restrict__ bias,
                          const float* __restrict__ w, float* __restrict__ y,
                          int Cin, int Cout, int Din, int Hin, int Win)
{
    const int Dout = Din * 2, Hout = Hin * 2, Wout = Win * 2;
    const int Nsp = Dout * Hout * Wout;
    const int tid = blockIdx.x * blockDim.x + threadIdx.x;
    if (tid >= Nsp * Cout) return;
    const int co = tid / Nsp;
    const int m  = tid - co * Nsp;
    const int od = m / (Hout * Wout);
    const int r  = m - od * Hout * Wout;
    const int oh = r / Wout, ow = r - oh * Wout;
    const float b = *bias;

    float acc = 0.f;
    for (int ci = 0; ci < Cin; ++ci) {
        const float* xbp = x + ci * Din * Hin * Win;
        const float* wb  = w + (ci * Cout + co) * 64;   // w: (Cin, Cout, 4,4,4)
        for (int kd = (od + 1) & 1; kd < 4; kd += 2) {
            int id = (od + 1 - kd) >> 1;
            if ((unsigned)id >= (unsigned)Din) continue;
            for (int kh = (oh + 1) & 1; kh < 4; kh += 2) {
                int ih = (oh + 1 - kh) >> 1;
                if ((unsigned)ih >= (unsigned)Hin) continue;
                for (int kw = (ow + 1) & 1; kw < 4; kw += 2) {
                    int iw = (ow + 1 - kw) >> 1;
                    if ((unsigned)iw >= (unsigned)Win) continue;
                    acc += (xbp[(id * Hin + ih) * Win + iw] + b) *
                           wb[(kd * 4 + kh) * 4 + kw];
                }
            }
        }
    }
    y[tid] = acc;
}

// ---------------------------------------------------------------------------
// y = nca(leaky(x + bias)) : shifted 2x2x2 average (pad-low-by-1); padded
// zeros stay raw zero (pad applied AFTER leaky in the reference).
// ---------------------------------------------------------------------------
__global__ void k_leaky_nca(const float* __restrict__ x, const float* __restrict__ bias,
                            float* __restrict__ y, int C, int D, int H, int W)
{
    const int n = C * D * H * W;
    const int t = blockIdx.x * blockDim.x + threadIdx.x;
    if (t >= n) return;
    const int w0 = t % W; int r = t / W;
    const int h0 = r % H; r /= H;
    const int d0 = r % D; const int c = r / D;
    const float b = *bias;
    const float* xb = x + c * D * H * W;
    float s = 0.f;
#pragma unroll
    for (int dz = -1; dz <= 0; ++dz)
#pragma unroll
        for (int hz = -1; hz <= 0; ++hz)
#pragma unroll
            for (int wz = -1; wz <= 0; ++wz) {
                int dd = d0 + dz, hh = h0 + hz, ww = w0 + wz;
                if ((unsigned)dd < (unsigned)D && (unsigned)hh < (unsigned)H &&
                    (unsigned)ww < (unsigned)W)
                    s += leaky1(xb[(dd * H + hh) * W + ww] + b);
            }
    y[t] = 0.125f * s;
}

// y = leaky(c2*scale + b2b + nca(skip))
__global__ void k_block_out(const float* __restrict__ c2, const float* __restrict__ scale,
                            const float* __restrict__ b2b, const float* __restrict__ sk,
                            float* __restrict__ y, int C, int D, int H, int W)
{
    const int n = C * D * H * W;
    const int t = blockIdx.x * blockDim.x + threadIdx.x;
    if (t >= n) return;
    const int w0 = t % W; int r = t / W;
    const int h0 = r % H; r /= H;
    const int d0 = r % D; const int c = r / D;
    const float* sb = sk + c * D * H * W;
    float s = 0.f;
#pragma unroll
    for (int dz = -1; dz <= 0; ++dz)
#pragma unroll
        for (int hz = -1; hz <= 0; ++hz)
#pragma unroll
            for (int wz = -1; wz <= 0; ++wz) {
                int dd = d0 + dz, hh = h0 + hz, ww = w0 + wz;
                if ((unsigned)dd < (unsigned)D && (unsigned)hh < (unsigned)H &&
                    (unsigned)ww < (unsigned)W)
                    s += sb[(dd * H + hh) * W + ww];
            }
    y[t] = leaky1(c2[t] * (*scale) + (*b2b) + 0.125f * s);
}

// ---------------------------------------------------------------------------
// VQ: dists[n][k] = |x_n|^2 + |e_k|^2 - 2 x_n.e_k, argmin over 256 codes.
// Dot products via WMMA (K = C channels padded to 32).  Forward value of the
// straight-through output is exactly emb[idx].
// ---------------------------------------------------------------------------
__global__ void k_vq_wmma(const float* __restrict__ x,   // [C, N]
                          const float* __restrict__ emb, // [256, C]
                          float* __restrict__ q,         // [C, N]
                          int C, int N,
                          float* __restrict__ loss_acc, unsigned int* __restrict__ hist)
{
    __shared__ float lds_dot[16][17];
    __shared__ float lds_en[16];

    const int lane = threadIdx.x & 31;
    const int hl   = lane >> 4;
    const int mr   = lane & 15;
    const int n0   = blockIdx.x * 16;
    const int vox  = n0 + mr;
    const bool mval = (vox < N);

    v16h a;
#pragma unroll
    for (int e = 0; e < 16; ++e) {
        int kg = kmapA(e, hl);
        float av = (mval && kg < C) ? x[kg * N + vox] : 0.f;
        a[e] = (_Float16)av;
    }

    float xn = 0.f;
    if (lane < 16 && mval)
        for (int c = 0; c < C; ++c) { float v = x[c * N + vox]; xn += v * v; }

    float best = 3.4e38f; int bidx = 0;
    for (int cc = 0; cc < 256; cc += 16) {
        if (lane < 16) {
            float en = 0.f;
            for (int c = 0; c < C; ++c) { float v = emb[(cc + lane) * C + c]; en += v * v; }
            lds_en[lane] = en;
        }
        v16h bm;
#pragma unroll
        for (int e = 0; e < 16; ++e) {
            int kg = kmapB(e, hl);
            float bv = (kg < C) ? emb[(cc + (lane & 15)) * C + kg] : 0.f;
            bm[e] = (_Float16)bv;
        }
        v8f dots = {};
        dots = __builtin_amdgcn_wmma_f32_16x16x32_f16(false, a, false, bm,
                                                      (short)0, dots, false, false);
#pragma unroll
        for (int v = 0; v < 8; ++v) lds_dot[v + hl * 8][lane & 15] = dots[v];
        __syncthreads();
        if (lane < 16 && mval) {
            for (int j = 0; j < 16; ++j) {
                float d = xn + lds_en[j] - 2.f * lds_dot[mr][j];
                if (d < best) { best = d; bidx = cc + j; }
            }
        }
        __syncthreads();
    }

    if (lane < 16 && mval) {
        float sq = 0.f;
        for (int c = 0; c < C; ++c) {
            float e  = emb[bidx * C + c];
            float xv = x[c * N + vox];
            q[c * N + vox] = e;
            float d = e - xv; sq += d * d;
        }
        atomicAdd(loss_acc, sq);
        atomicAdd(&hist[bidx], 1u);
    }
}

__global__ void k_copy(const float* __restrict__ s, float* __restrict__ d, int n)
{
    int t = blockIdx.x * blockDim.x + threadIdx.x;
    if (t < n) d[t] = s[t];
}

// loss = sum_l COMMIT * acc_l/(N_l*C_l);  perplexity_l = exp(-sum p log(p+1e-10))
__global__ void k_finalize(const float* __restrict__ acc, const unsigned int* __restrict__ hist,
                           float* __restrict__ out)
{
    if (threadIdx.x | blockIdx.x) return;
    const int Ns[3] = {8, 512, 32768};
    const int Cs[3] = {32, 8, 2};
    float loss = 0.f;
    for (int l = 0; l < 3; ++l) loss += COMMIT * acc[l] / (float)(Ns[l] * Cs[l]);
    out[0] = loss;
    for (int l = 0; l < 3; ++l) {
        float s = 0.f;
        for (int k = 0; k < 256; ++k) {
            float p = (float)hist[l * 256 + k] / (float)Ns[l];
            s += p * logf(p + 1e-10f);
        }
        out[1 + l] = expf(-s);
    }
}

// ---------------------------------------------------------------------------
// Host orchestration
// ---------------------------------------------------------------------------
enum Kind { K_LEVEL = 0, K_DOWN = 1, K_UP = 2 };

static void run_conv(const float* x, const float* bias, const float* w, float* y,
                     int Cin, int Cout, int Ds, int stride, float* wpad, hipStream_t s)
{
    int wtotal = Cout * Cin * 32;
    k_pad_w<<<(unsigned)((wtotal + 255) / 256), 256, 0, s>>>(w, wpad, wtotal);
    int Do = Ds / stride;
    long Nsp = (long)Do * Do * Do;
    int nw = (int)((Nsp + 15) / 16); if (nw > 4) nw = 4;   // waves per block
    long mtiles = (Nsp + 16L * nw - 1) / (16L * nw);
    dim3 g((unsigned)mtiles, (unsigned)((Cout + 15) / 16));
    k_conv3d_wmma<<<g, nw * 32, 0, s>>>(x, bias, wpad, y, Cin, Cout,
                                        Ds, Ds, Ds, Do, Do, Do, stride);
}

static void run_fixup(void* const* din, int blk, int kind,
                      int Cin, int Cout, int Ds,
                      const float* in, float* out,
                      float* t1, float* t2, float* t3, float* wpad, hipStream_t s)
{
    // param order per block (setup_inputs insertion order):
    // conv1, skip, conv2, bias1a, bias1b, bias2a, bias2b, scale
    const int base = 1 + blk * 8;
    const float* conv1 = (const float*)din[base + 0];
    const float* skipw = (const float*)din[base + 1];
    const float* conv2 = (const float*)din[base + 2];
    const float* b1a   = (const float*)din[base + 3];
    const float* b1b   = (const float*)din[base + 4];
    const float* b2a   = (const float*)din[base + 5];
    const float* b2b   = (const float*)din[base + 6];
    const float* scale = (const float*)din[base + 7];

    const int Do = (kind == K_DOWN) ? Ds / 2 : (kind == K_UP ? Ds * 2 : Ds);
    const int Nout = Cout * Do * Do * Do;
    const unsigned gew = (unsigned)((Nout + 255) / 256);

    if (kind == K_UP)
        k_convT3d<<<gew, 256, 0, s>>>(in, b1a, conv1, t1, Cin, Cout, Ds, Ds, Ds);
    else
        run_conv(in, b1a, conv1, t1, Cin, Cout, Ds, kind == K_DOWN ? 2 : 1, wpad, s);

    k_leaky_nca<<<gew, 256, 0, s>>>(t1, b1b, t2, Cout, Do, Do, Do);
    run_conv(t2, b2a, conv2, t3, Cout, Cout, Do, 1, wpad, s);

    if (kind == K_UP)
        k_convT3d<<<gew, 256, 0, s>>>(in, b1a, skipw, t1, Cin, Cout, Ds, Ds, Ds);
    else
        run_conv(in, b1a, skipw, t1, Cin, Cout, Ds, kind == K_DOWN ? 2 : 1, wpad, s);

    k_block_out<<<gew, 256, 0, s>>>(t3, scale, b2b, t1, out, Cout, Do, Do, Do);
}

static void copyf(const float* src, float* dst, int n, hipStream_t s)
{
    k_copy<<<(unsigned)((n + 255) / 256), 256, 0, s>>>(src, dst, n);
}

extern "C" void kernel_launch(void* const* d_in, const int* in_sizes, int n_in,
                              void* d_out, int out_size, void* d_ws, size_t ws_size,
                              hipStream_t stream)
{
    (void)in_sizes; (void)n_in; (void)out_size; (void)ws_size;
    char* ws = (char*)d_ws;
    size_t off = 0;
    auto alloc = [&](size_t bytes) -> float* {
        float* p = (float*)(ws + off);
        off += (bytes + 255) & ~(size_t)255;
        return p;
    };
    const size_t SLAB = 34ull * 1024 * 1024;     // holds up to 4ch * 128^3 fp32
    float* P0 = alloc(SLAB);
    float* P1 = alloc(SLAB);
    float* T1 = alloc(SLAB);
    float* T2 = alloc(SLAB);
    float* T3 = alloc(SLAB);
    float* WP  = alloc(256ull * 256 * 32 * 4);   // padded weights (max: 256x256 conv2)
    float* e2p = alloc(16ull * 32768 * 4);
    float* e4p = alloc(64ull * 512 * 4);
    float* q6p = alloc(32ull * 8 * 4);
    float* q4p = alloc(8ull * 512 * 4);
    float* q2p = alloc(2ull * 32768 * 4);
    float* acc = alloc(3 * 4);
    unsigned int* hist = (unsigned int*)alloc(3 * 256 * 4);

    hipMemsetAsync(acc, 0, 3 * 4, stream);
    hipMemsetAsync(hist, 0, 3 * 256 * 4, stream);

    const float* x  = (const float*)d_in[0];
    const float* E6 = (const float*)d_in[1 + 21 * 8 + 0];
    const float* E4 = (const float*)d_in[1 + 21 * 8 + 1];
    const float* E2 = (const float*)d_in[1 + 21 * 8 + 2];

    // ---- Encoder ----
    run_fixup(d_in, 0, K_LEVEL, 1, 4, 128, x,  P0, T1, T2, T3, WP, stream);  // e0 [4,128]
    run_fixup(d_in, 1, K_DOWN,  4, 8, 128, P0, P1, T1, T2, T3, WP, stream);  // e1 [8,64]
    run_fixup(d_in, 2, K_DOWN,  8, 16, 64, P1, P0, T1, T2, T3, WP, stream);  // e2 [16,32]
    copyf(P0, e2p, 16 * 32768, stream);
    run_fixup(d_in, 3, K_DOWN, 16, 32, 32, P0, P1, T1, T2, T3, WP, stream);  // e3 [32,16]
    run_fixup(d_in, 4, K_DOWN, 32, 64, 16, P1, P0, T1, T2, T3, WP, stream);  // e4 [64,8]
    copyf(P0, e4p, 64 * 512, stream);
    run_fixup(d_in, 5, K_DOWN, 64, 128, 8, P0, P1, T1, T2, T3, WP, stream);  // e5 [128,4]
    run_fixup(d_in, 6, K_DOWN, 128, 256, 4, P1, P0, T1, T2, T3, WP, stream); // e6 [256,2]

    // ---- Quantization ----
    run_fixup(d_in, 7, K_LEVEL, 256, 32, 2, P0, P1, T1, T2, T3, WP, stream); // q6b [32,2]
    k_vq_wmma<<<1, 32, 0, stream>>>(P1, E6, q6p, 32, 8, acc + 0, hist + 0);
    run_fixup(d_in, 8, K_UP, 32, 16, 2, q6p, P0, T1, T2, T3, WP, stream);    // [16,4]
    run_fixup(d_in, 9, K_UP, 16, 8, 4, P0, P1, T1, T2, T3, WP, stream);      // [8,8]
    copyf(e4p, P0, 64 * 512, stream);
    copyf(P1, P0 + 64 * 512, 8 * 512, stream);                               // concat [72,8]
    run_fixup(d_in, 10, K_LEVEL, 72, 8, 8, P0, P1, T1, T2, T3, WP, stream);  // q4b [8,8]
    k_vq_wmma<<<32, 32, 0, stream>>>(P1, E4, q4p, 8, 512, acc + 1, hist + 256);
    run_fixup(d_in, 11, K_UP, 8, 4, 8, q4p, P0, T1, T2, T3, WP, stream);     // [4,16]
    run_fixup(d_in, 12, K_UP, 4, 2, 16, P0, P1, T1, T2, T3, WP, stream);     // [2,32]
    copyf(e2p, P0, 16 * 32768, stream);
    copyf(P1, P0 + 16 * 32768, 2 * 32768, stream);                           // concat [18,32]
    run_fixup(d_in, 13, K_LEVEL, 18, 2, 32, P0, P1, T1, T2, T3, WP, stream); // q2b [2,32]
    k_vq_wmma<<<2048, 32, 0, stream>>>(P1, E2, q2p, 2, 32768, acc + 2, hist + 512);

    // ---- Decoder ----
    run_fixup(d_in, 14, K_UP, 32, 128, 2, q6p, P0, T1, T2, T3, WP, stream);  // d6 [128,4]
    run_fixup(d_in, 15, K_UP, 128, 64, 4, P0, P1, T1, T2, T3, WP, stream);   // d5 [64,8]
    copyf(P1, P0, 64 * 512, stream);
    copyf(q4p, P0 + 64 * 512, 8 * 512, stream);                              // concat [72,8]
    run_fixup(d_in, 16, K_UP, 72, 32, 8, P0, P1, T1, T2, T3, WP, stream);    // d4 [32,16]
    run_fixup(d_in, 17, K_UP, 32, 16, 16, P1, P0, T1, T2, T3, WP, stream);   // d3 [16,32]
    copyf(P0, P1, 16 * 32768, stream);
    copyf(q2p, P1 + 16 * 32768, 2 * 32768, stream);                          // concat [18,32]
    run_fixup(d_in, 18, K_UP, 18, 8, 32, P1, P0, T1, T2, T3, WP, stream);    // d2 [8,64]
    run_fixup(d_in, 19, K_UP, 8, 4, 64, P0, P1, T1, T2, T3, WP, stream);     // d1 [4,128]
    run_fixup(d_in, 20, K_LEVEL, 4, 1, 128, P1, (float*)d_out, T1, T2, T3, WP, stream); // recon

    k_finalize<<<1, 1, 0, stream>>>(acc, hist, (float*)d_out + 2097152);
}
